// LiteDeformConv_41068477284906
// MI455X (gfx1250) — matrix operand order repaired and checked by reading, not compiled, not run on hardware
//
#include <hip/hip_runtime.h>
#include <math.h>

// ---------------------------------------------------------------------------
// CDNA5 (gfx1250) wave32 WMMA implementation of LiteDeformConv decoder.
// All GEMM-shaped work (1x1 convs, deformable-conv einsum, transposed conv,
// offset conv, final 3x3 conv) runs through one bf16 WMMA GEMM kernel
// (v_wmma_f32_16x16x32_bf16) with fused epilogues.
// Tile fills are software-pipelined: next K-tile's global loads are issued
// before the current tile's WMMA work, with register staging, so the
// s_wait_loadcnt lands after a full tile of matrix math.
// fp32->bf16 uses packed hardware converts (v_cvt_pk_bf16_f32).
// ---------------------------------------------------------------------------

typedef __attribute__((ext_vector_type(16))) __bf16    v16bf;
typedef __attribute__((ext_vector_type(2)))  __bf16    v2bf;
typedef __attribute__((ext_vector_type(8)))  float     v8f;
typedef __attribute__((ext_vector_type(4)))  float     v4f;
typedef __attribute__((ext_vector_type(2)))  float     v2f;
typedef __attribute__((ext_vector_type(4)))  unsigned  v4u;
typedef __attribute__((ext_vector_type(2)))  unsigned  v2u;

#define TB 64   // block tile (M and N)
#define KT 64   // K tile = two WMMA k-steps per barrier pair
#define LP 72   // LDS row pitch in shorts (144 B, 16B-aligned rows)

enum { B_DENSE = 0, B_IM2COL3 = 1 };
enum { EPI_NONE = 0, EPI_BIAS = 1, EPI_BIAS_RES = 2, EPI_BN_RELU = 3, EPI_BN_RELU_PAR = 4 };

// native RNE converts (hardware bf16 cvt on gfx1250)
__device__ __forceinline__ unsigned short f2bf(float f) {
  __bf16 b = (__bf16)f;
  return __builtin_bit_cast(unsigned short, b);
}
// packed pair conversion -> single v_cvt_pk_bf16_f32
__device__ __forceinline__ unsigned pk2(float a, float b) {
  v2f  f = { a, b };
  v2bf p = __builtin_convertvector(f, v2bf);
  return __builtin_bit_cast(unsigned, p);
}

// ---------------------------------------------------------------------------
// WMMA GEMM:  C[M,N] = A[M,K] * B[K,N]  (+ epilogue)
//   bmode = B_DENSE  : B is [K,N] row-major
//   bmode = B_IM2COL3: B row k = c*9 + (ky*3+kx); value = x[c, h+ky-1, w+kx-1]
// Requires: N % 64 == 0, K % 64 == 0 (true at all call sites). M arbitrary.
// Block: 128 threads = 4 waves; each wave owns a 32x32 accumulator quad.
// ---------------------------------------------------------------------------
__global__ void __launch_bounds__(128)
wmma_gemm_kernel(const float* __restrict__ A,
                 const float* __restrict__ Bsrc,
                 float* __restrict__ Cout,
                 int M, int N, int K,
                 int bmode, int H, int W,
                 int epi,
                 const float* __restrict__ p0,
                 const float* __restrict__ p1,
                 const float* __restrict__ res,
                 int ldc, int outW2, int py, int px)
{
  __shared__ unsigned short As[TB][LP];   // [m][k]
  __shared__ unsigned short Bs[TB][LP];   // [n][k] (transposed on fill)

  const int tid  = threadIdx.x;
  const int lane = tid & 31;
  const int wid  = tid >> 5;
  const int blockM = blockIdx.y * TB;
  const int blockN = blockIdx.x * TB;
  const int waveM = (wid >> 1) * 32;
  const int waveN = (wid & 1) * 32;
  const int HW = H * W;
  const bool dense = (bmode == B_DENSE);

  v8f acc[2][2] = {};

  // ---- loop-invariant fill geometry (A row ptr + M-mask independent of k0)
  const float* aptr[8];
  float amask[8];
  #pragma unroll
  for (int i = 0; i < 8; ++i) {
    int e = tid + i * 128;               // float4 slot: 64 rows x 16 quads
    int m = e >> 4, c4 = e & 15;
    int gm = blockM + m;
    aptr[i]  = A + (long)min(gm, M - 1) * K + c4 * 4;
    amask[i] = (gm < M) ? 1.f : 0.f;
  }

  // ---- staging registers ----
  v4f   ra[8];     // A tile
  v4f   rb[8];     // B tile (dense): 2 chunks x 4 k-rows
  float rbi[32];   // B tile (im2col)

  auto loadA = [&](int k0) {
    #pragma unroll
    for (int i = 0; i < 8; ++i) ra[i] = *(const v4f*)(aptr[i] + k0);
  };
  auto storeA = [&]() {
    #pragma unroll
    for (int i = 0; i < 8; ++i) {
      int e = tid + i * 128;
      int m = e >> 4, c4 = e & 15;
      v4f x = ra[i] * amask[i];
      v2u pkd = { pk2(x[0], x[1]), pk2(x[2], x[3]) };
      *(v2u*)&As[m][c4 * 4] = pkd;
    }
  };
  auto loadB = [&](int k0) {
    #pragma unroll
    for (int i = 0; i < 2; ++i) {
      int e  = tid + i * 128;            // chunk: 16 k-groups x 16 n-groups
      int kg = e >> 4, ng = e & 15;
      const float* bp = Bsrc + (long)(k0 + kg * 4) * N + (blockN + ng * 4);
      rb[i * 4 + 0] = *(const v4f*)(bp);
      rb[i * 4 + 1] = *(const v4f*)(bp + (long)N);
      rb[i * 4 + 2] = *(const v4f*)(bp + 2l * N);
      rb[i * 4 + 3] = *(const v4f*)(bp + 3l * N);
    }
  };
  auto storeB = [&]() {
    #pragma unroll
    for (int i = 0; i < 2; ++i) {
      int e  = tid + i * 128;
      int kg = e >> 4, ng = e & 15;
      #pragma unroll
      for (int j = 0; j < 4; ++j) {
        v2u pkd = { pk2(rb[i * 4 + 0][j], rb[i * 4 + 1][j]),
                    pk2(rb[i * 4 + 2][j], rb[i * 4 + 3][j]) };
        *(v2u*)&Bs[ng * 4 + j][kg * 4] = pkd;
      }
    }
  };
  auto loadBi = [&](int k0) {            // branchless clamped im2col gather
    #pragma unroll 8
    for (int i = 0; i < 32; ++i) {
      int e  = i * 128 + tid;
      int n  = e & 63, kk = e >> 6;
      int gn = blockN + n, gk = k0 + kk;
      int c  = gk / 9, t = gk - 9 * c;
      int ky = t / 3 - 1, kx = (t % 3) - 1;
      int h  = gn / W, w = gn - h * W;
      int hh = h + ky, ww = w + kx;
      bool ok = ((unsigned)hh < (unsigned)H) && ((unsigned)ww < (unsigned)W);
      int hc = min(max(hh, 0), H - 1), wc = min(max(ww, 0), W - 1);
      float v = Bsrc[(long)c * HW + hc * W + wc];   // always in-bounds
      rbi[i] = ok ? v : 0.f;
    }
  };
  auto storeBi = [&]() {
    #pragma unroll 8
    for (int i = 0; i < 32; ++i) {
      int e = i * 128 + tid;
      int n = e & 63, kk = e >> 6;
      Bs[n][kk] = f2bf(rbi[i]);
    }
  };

  // ---- software pipeline: prologue loads for tile 0 ----
  loadA(0);
  if (dense) loadB(0); else loadBi(0);

  for (int k0 = 0; k0 < K; k0 += KT) {
    // flush staged tile to LDS (waits on its global loads here)
    storeA();
    if (dense) storeB(); else storeBi();
    __syncthreads();

    // issue next tile's global loads NOW; they overlap all WMMA work below
    int k1 = k0 + KT;
    if (k1 < K) {
      loadA(k1);
      if (dense) loadB(k1); else loadBi(k1);
    }

    // ---- fragment assembly per CDNA5 16-bit WMMA layouts; 2 k-steps/tile
    const int l16 = lane & 15;
    const int hi  = lane >> 4;
    #pragma unroll
    for (int ks = 0; ks < 2; ++ks) {
      v16bf af[2], bg[2];
      #pragma unroll
      for (int mi = 0; mi < 2; ++mi) {
        const unsigned short* r = &As[waveM + mi * 16 + l16][ks * 32];
        // A: lane<16 -> K {0..7,16..23}; lane>=16 -> K {8..15,24..31}
        ((v4u*)&af[mi])[0] = *(const v4u*)(r + hi * 8);
        ((v4u*)&af[mi])[1] = *(const v4u*)(r + hi * 8 + 16);
      }
      #pragma unroll
      for (int ni = 0; ni < 2; ++ni) {
        const unsigned short* r = &Bs[waveN + ni * 16 + l16][ks * 32];
        // B: lane<16 -> K {0..7,16..23}; lane>=16 -> K {8..15,24..31} (N = lane&15)
        ((v4u*)&bg[ni])[0] = *(const v4u*)(r + hi * 8);
        ((v4u*)&bg[ni])[1] = *(const v4u*)(r + hi * 8 + 16);
      }
      #pragma unroll
      for (int mi = 0; mi < 2; ++mi)
        #pragma unroll
        for (int ni = 0; ni < 2; ++ni)
          acc[mi][ni] = __builtin_amdgcn_wmma_f32_16x16x32_bf16(
              false, af[mi], false, bg[ni], (short)0, acc[mi][ni], false, false);
    }
    __syncthreads();
  }

  // ---- epilogue + store: C element (m,n): lane = (n&15)+16*(m>=8), vgpr = m%8
  #pragma unroll
  for (int mi = 0; mi < 2; ++mi) {
    #pragma unroll
    for (int ni = 0; ni < 2; ++ni) {
      #pragma unroll
      for (int i = 0; i < 8; ++i) {
        int gm = blockM + waveM + mi * 16 + (lane >> 4) * 8 + i;
        int gn = blockN + waveN + ni * 16 + (lane & 15);
        if (gm < M) {                                   // N always tile-aligned
          float v = acc[mi][ni][i];
          if (epi == EPI_BIAS)          v += p0[gm];
          else if (epi == EPI_BIAS_RES) v += p0[gm] + res[(long)gm * N + gn];
          else if (epi >= EPI_BN_RELU) { v = fmaxf(v * p0[gm] + p1[gm], 0.0f); }
          long ci;
          if (epi == EPI_BN_RELU_PAR) {
            int r = gn / W, cc = gn - r * W;
            ci = (long)gm * ldc + (2 * r + py) * outW2 + (2 * cc + px);
          } else {
            ci = (long)gm * ldc + gn;
          }
          Cout[ci] = v;
        }
      }
    }
  }
}

// ---------------------------------------------------------------------------
// BN folding: scale = g/sqrt(v+eps), shift = b - m*scale
// ---------------------------------------------------------------------------
__global__ void bn_prep_kernel(const float* __restrict__ g, const float* __restrict__ b,
                               const float* __restrict__ m, const float* __restrict__ v,
                               float* __restrict__ scale, float* __restrict__ shift, int C)
{
  int i = blockIdx.x * blockDim.x + threadIdx.x;
  if (i < C) {
    float s = g[i] * rsqrtf(v[i] + 1e-5f);
    scale[i] = s;
    shift[i] = b[i] - m[i] * s;
  }
}

// ---------------------------------------------------------------------------
// Modulated deformable-conv sampler -> samp[c*9+k, p] (one batch).
// om layout: dy_k = om[2k], dx_k = om[2k+1], mask_k = sigmoid(om[18+k]).
// ---------------------------------------------------------------------------
__global__ void mdcn_sample_kernel(const float* __restrict__ x,   // [C,H,W]
                                   const float* __restrict__ om,  // [27,H,W]
                                   float* __restrict__ samp,      // [C*9, H*W]
                                   int C, int H, int W)
{
  int HW = H * W;
  int k  = blockIdx.y;
  int p  = blockIdx.x * blockDim.x + threadIdx.x;
  if (p >= HW) return;
  int h = p / W, w = p - h * W;

  float dy = om[(long)(2 * k) * HW + p];
  float dx = om[(long)(2 * k + 1) * HW + p];
  float mk = 1.0f / (1.0f + __expf(-om[(long)(18 + k) * HW + p]));

  int   ky = k / 3, kx = k % 3;
  float pyf = (float)h + (float)(ky - 1) + dy;
  float pxf = (float)w + (float)(kx - 1) + dx;
  float y0f = floorf(pyf), x0f = floorf(pxf);
  float wy = pyf - y0f, wx = pxf - x0f;
  int y0 = (int)y0f, x0 = (int)x0f, y1 = y0 + 1, x1 = x0 + 1;
  float vy0 = (y0 >= 0 && y0 < H) ? 1.f : 0.f;
  float vy1 = (y1 >= 0 && y1 < H) ? 1.f : 0.f;
  float vx0 = (x0 >= 0 && x0 < W) ? 1.f : 0.f;
  float vx1 = (x1 >= 0 && x1 < W) ? 1.f : 0.f;
  int y0c = min(max(y0, 0), H - 1), y1c = min(max(y1, 0), H - 1);
  int x0c = min(max(x0, 0), W - 1), x1c = min(max(x1, 0), W - 1);
  float w00 = (1.f - wy) * (1.f - wx) * vy0 * vx0 * mk;
  float w01 = (1.f - wy) * wx         * vy0 * vx1 * mk;
  float w10 = wy * (1.f - wx)         * vy1 * vx0 * mk;
  float w11 = wy * wx                 * vy1 * vx1 * mk;
  int i00 = y0c * W + x0c, i01 = y0c * W + x1c;
  int i10 = y1c * W + x0c, i11 = y1c * W + x1c;

  for (int c = 0; c < C; ++c) {
    const float* xb = x + (long)c * HW;
    float v = w00 * xb[i00] + w01 * xb[i01] + w10 * xb[i10] + w11 * xb[i11];
    samp[(long)(c * 9 + k) * HW + p] = v;
  }
}

// ---------------------------------------------------------------------------
// Transposed-conv (k=4,s=2,p=1) decomposed into 4 output parities:
//   y[o,2r+py,2c+px] = sum_i sum_{ty,tx} x[i, m(ty), n(tx)] * w[i,o,ky,kx]
//   ty=0: m=r,      ky=1+py ;  ty=1: m=r-1+2py, ky=3-3py   (same for x-dim)
// ---------------------------------------------------------------------------
__global__ void convt_pack_kernel(const float* __restrict__ upw,  // [Cin][Cout][4][4]
                                  float* __restrict__ Wp,         // [4][Cout][Cin*4]
                                  int Cin, int Cout)
{
  long total = (long)4 * Cout * Cin * 4;
  long e = (long)blockIdx.x * blockDim.x + threadIdx.x;
  if (e >= total) return;
  int  K4     = Cin * 4;
  int  parity = (int)(e / ((long)Cout * K4));
  long rem    = e % ((long)Cout * K4);
  int  o  = (int)(rem / K4);
  int  kk = (int)(rem % K4);
  int  i  = kk >> 2, t = kk & 3, ty = t >> 1, tx = t & 1;
  int  py = parity >> 1, px = parity & 1;
  int  ky = ty ? (3 - 3 * py) : (1 + py);
  int  kx = tx ? (3 - 3 * px) : (1 + px);
  Wp[e] = upw[(((long)i * Cout + o) * 4 + ky) * 4 + kx];
}

__global__ void convt_col_kernel(const float* __restrict__ x,  // [C,H,W]
                                 float* __restrict__ col,      // [C*4, H*W]
                                 int C, int H, int W, int py, int px)
{
  long total = (long)C * 4 * H * W;
  long e = (long)blockIdx.x * blockDim.x + threadIdx.x;
  if (e >= total) return;
  int HW  = H * W;
  int p   = (int)(e % HW);
  int row = (int)(e / HW);
  int i = row >> 2, t = row & 3, ty = t >> 1, tx = t & 1;
  int r = p / W, cc = p - r * W;
  int m = ty ? (r - 1 + 2 * py) : r;
  int n = tx ? (cc - 1 + 2 * px) : cc;
  bool ok = ((unsigned)m < (unsigned)H) && ((unsigned)n < (unsigned)W);
  int mc = min(max(m, 0), H - 1), nc = min(max(n, 0), W - 1);
  float v = x[(long)i * HW + mc * W + nc];
  col[e] = ok ? v : 0.f;
}

// ---------------------------------------------------------------------------
// y (in place, holds z2) += bias[ch] + up2(z3) + up4(z4) + up8(z5)
// half-pixel-center bilinear with edge clamp (jax.image.resize 'bilinear')
// ---------------------------------------------------------------------------
__device__ __forceinline__ float bilerp(const float* __restrict__ z, int H, int W,
                                        float sy, float sx)
{
  float y0f = floorf(sy), x0f = floorf(sx);
  int y0 = (int)y0f, x0 = (int)x0f;
  float fy = sy - y0f, fx = sx - x0f;
  int y0c = min(max(y0, 0), H - 1), y1c = min(max(y0 + 1, 0), H - 1);
  int x0c = min(max(x0, 0), W - 1), x1c = min(max(x0 + 1, 0), W - 1);
  float v00 = z[y0c * W + x0c], v01 = z[y0c * W + x1c];
  float v10 = z[y1c * W + x0c], v11 = z[y1c * W + x1c];
  return (1.f - fy) * ((1.f - fx) * v00 + fx * v01) +
         fy        * ((1.f - fx) * v10 + fx * v11);
}

__global__ void fuse_y_kernel(float* __restrict__ y,
                              const float* __restrict__ z3,
                              const float* __restrict__ z4,
                              const float* __restrict__ z5,
                              const float* __restrict__ bias,
                              int CH, int HO, int WO,
                              int H3, int W3, int H4, int W4, int H5, int W5)
{
  int tot = CH * HO * WO;
  int e = blockIdx.x * blockDim.x + threadIdx.x;
  if (e >= tot) return;
  int ch  = e / (HO * WO);
  int pix = e - ch * HO * WO;
  int oy = pix / WO, ox = pix - oy * WO;
  float sy3 = (oy + 0.5f) * ((float)H3 / HO) - 0.5f;
  float sx3 = (ox + 0.5f) * ((float)W3 / WO) - 0.5f;
  float sy4 = (oy + 0.5f) * ((float)H4 / HO) - 0.5f;
  float sx4 = (ox + 0.5f) * ((float)W4 / WO) - 0.5f;
  float sy5 = (oy + 0.5f) * ((float)H5 / HO) - 0.5f;
  float sx5 = (ox + 0.5f) * ((float)W5 / WO) - 0.5f;
  float v = y[e] + bias[ch]
          + bilerp(z3 + (long)ch * H3 * W3, H3, W3, sy3, sx3)
          + bilerp(z4 + (long)ch * H4 * W4, H4, W4, sy4, sx4)
          + bilerp(z5 + (long)ch * H5 * W5, H5, W5, sy5, sx5);
  y[e] = v;
}

// ---------------------------------------------------------------------------
// Host-side orchestration
// ---------------------------------------------------------------------------
static inline void launch_gemm(hipStream_t st, const float* A, const float* B, float* C,
                               int M, int N, int K, int bmode, int H, int W, int epi,
                               const float* p0, const float* p1, const float* res,
                               int ldc, int outW2 = 0, int py = 0, int px = 0)
{
  dim3 g((N + TB - 1) / TB, (M + TB - 1) / TB);
  wmma_gemm_kernel<<<g, 128, 0, st>>>(A, B, C, M, N, K, bmode, H, W,
                                      epi, p0, p1, res, ldc, outW2, py, px);
}

struct DLp {
  const float *off_w, *off_b, *dcn_w, *up_w;
  const float *g1, *b1, *m1, *v1, *g2, *b2, *m2, *v2;
};

static void deform_layer(hipStream_t st, const float* x, int Cin, int Cout, int H, int W,
                         const DLp& p,
                         float* sc1, float* sh1, float* sc2, float* sh2,
                         float* om, float* samp, float* col, float* Wp,
                         float* dcnb, float* upout)
{
  int HW = H * W;
  bn_prep_kernel<<<(Cout + 255) / 256, 256, 0, st>>>(p.g1, p.b1, p.m1, p.v1, sc1, sh1, Cout);
  bn_prep_kernel<<<(Cout + 255) / 256, 256, 0, st>>>(p.g2, p.b2, p.m2, p.v2, sc2, sh2, Cout);
  long wtot = (long)4 * Cout * Cout * 4;
  convt_pack_kernel<<<(int)((wtot + 255) / 256), 256, 0, st>>>(p.up_w, Wp, Cout, Cout);

  for (int b = 0; b < 2; ++b) {
    const float* xb = x + (long)b * Cin * HW;
    // offset conv (27 ch, 3x3 pad1) via implicit-im2col WMMA GEMM
    launch_gemm(st, p.off_w, xb, om, 27, HW, Cin * 9, B_IM2COL3, H, W,
                EPI_BIAS, p.off_b, nullptr, nullptr, HW);
    // deformable bilinear sampling -> samp[Cin*9, HW]
    dim3 sg((HW + 255) / 256, 9);
    mdcn_sample_kernel<<<sg, 256, 0, st>>>(xb, om, samp, Cin, H, W);
    // dcn einsum GEMM + fused BN1 + ReLU
    launch_gemm(st, p.dcn_w, samp, dcnb, Cout, HW, Cin * 9, B_DENSE, H, W,
                EPI_BN_RELU, sc1, sh1, nullptr, HW);
    // transposed conv (4 parity GEMMs) + fused BN2 + ReLU, strided store
    float* upb = upout + (long)b * Cout * 4 * HW;
    for (int par = 0; par < 4; ++par) {
      int py = par >> 1, px = par & 1;
      long ct = (long)Cout * 4 * HW;
      convt_col_kernel<<<(int)((ct + 255) / 256), 256, 0, st>>>(dcnb, col, Cout, H, W, py, px);
      launch_gemm(st, Wp + (long)par * Cout * Cout * 4, col, upb, Cout, HW, Cout * 4,
                  B_DENSE, H, W, EPI_BN_RELU_PAR, sc2, sh2, nullptr, 4 * HW, 2 * W, py, px);
    }
  }
}

extern "C" void kernel_launch(void* const* d_in, const int* in_sizes, int n_in,
                              void* d_out, int out_size, void* d_ws, size_t ws_size,
                              hipStream_t stream)
{
  (void)in_sizes; (void)n_in; (void)out_size; (void)ws_size;
  // ---- inputs (setup_inputs() flattening order) ----
  const float* f2 = (const float*)d_in[0];   // (2,256,128,128)
  const float* f3 = (const float*)d_in[1];   // (2,512,64,64)
  const float* f4 = (const float*)d_in[2];   // (2,1024,32,32)
  const float* f5 = (const float*)d_in[3];   // (2,2048,16,16)
  const float* lat0_w = (const float*)d_in[4];
  const float* lat0_b = (const float*)d_in[5];
  DLp dl1 = { (const float*)d_in[6],  (const float*)d_in[7],  (const float*)d_in[8],
              (const float*)d_in[13],
              (const float*)d_in[9],  (const float*)d_in[10], (const float*)d_in[11], (const float*)d_in[12],
              (const float*)d_in[14], (const float*)d_in[15], (const float*)d_in[16], (const float*)d_in[17] };
  const float* lat1_w = (const float*)d_in[18];
  const float* lat1_b = (const float*)d_in[19];
  DLp dl2 = { (const float*)d_in[20], (const float*)d_in[21], (const float*)d_in[22],
              (const float*)d_in[27],
              (const float*)d_in[23], (const float*)d_in[24], (const float*)d_in[25], (const float*)d_in[26],
              (const float*)d_in[28], (const float*)d_in[29], (const float*)d_in[30], (const float*)d_in[31] };
  const float* lat2_w = (const float*)d_in[32];
  const float* lat2_b = (const float*)d_in[33];
  DLp dl3 = { (const float*)d_in[34], (const float*)d_in[35], (const float*)d_in[36],
              (const float*)d_in[41],
              (const float*)d_in[37], (const float*)d_in[38], (const float*)d_in[39], (const float*)d_in[40],
              (const float*)d_in[42], (const float*)d_in[43], (const float*)d_in[44], (const float*)d_in[45] };
  const float* lat3_w = (const float*)d_in[46];
  const float* lat3_b = (const float*)d_in[47];
  const float* a5_w = (const float*)d_in[48];
  const float* a4_w = (const float*)d_in[49];
  const float* a3_w = (const float*)d_in[50];
  const float* a2_w = (const float*)d_in[51];
  const float* biasP = (const float*)d_in[52];
  const float* out_w = (const float*)d_in[53];
  const float* out_b = (const float*)d_in[54];
  float* out = (float*)d_out;

  // ---- workspace (bump allocator, floats) ----
  float* wp = (float*)d_ws;
  auto alloc = [&](size_t n) { float* r = wp; wp += n; return r; };
  float* x5    = alloc(2l * 1024 * 256);
  float* x4    = alloc(2l * 512 * 1024);
  float* x3    = alloc(2l * 256 * 4096);
  float* x2    = alloc(2l * 128 * 16384);
  float* upout = alloc(4194304);          // max deform-layer output (2,128,128,128)
  float* dcnb  = alloc(524288);           // per-batch dcn out, max (128,64,64)
  float* om    = alloc(110592);           // per-batch (27,64,64) max
  float* samp  = alloc(9437184);          // per-batch im2col max (2304,4096); col shares it
  float* Wp    = alloc(4194304);          // packed conv_t weights, max layer1
  float* z5    = alloc(2l * 128 * 256);
  float* z4    = alloc(2l * 128 * 1024);
  float* z3    = alloc(2l * 128 * 4096);
  float* ybuf  = alloc(2l * 128 * 16384); // holds z2 then fused y
  float* sc1 = alloc(512), *sh1 = alloc(512), *sc2 = alloc(512), *sh2 = alloc(512);
  float* col = samp;                      // reuse: samp dead before col is written

  // ---- x5 = lat0(f5) ----
  for (int b = 0; b < 2; ++b)
    launch_gemm(stream, lat0_w, f5 + (long)b * 2048 * 256, x5 + (long)b * 1024 * 256,
                1024, 256, 2048, B_DENSE, 16, 16, EPI_BIAS, lat0_b, nullptr, nullptr, 256);

  // ---- deform layer 1 (1024 -> 512, 16x16 -> 32x32) ----
  deform_layer(stream, x5, 1024, 512, 16, 16, dl1, sc1, sh1, sc2, sh2, om, samp, col, Wp, dcnb, upout);
  for (int b = 0; b < 2; ++b)
    launch_gemm(stream, lat1_w, f4 + (long)b * 1024 * 1024, x4 + (long)b * 512 * 1024,
                512, 1024, 1024, B_DENSE, 32, 32, EPI_BIAS_RES, lat1_b, nullptr,
                upout + (long)b * 512 * 1024, 1024);

  // ---- deform layer 2 (512 -> 256, 32x32 -> 64x64) ----
  deform_layer(stream, x4, 512, 256, 32, 32, dl2, sc1, sh1, sc2, sh2, om, samp, col, Wp, dcnb, upout);
  for (int b = 0; b < 2; ++b)
    launch_gemm(stream, lat2_w, f3 + (long)b * 512 * 4096, x3 + (long)b * 256 * 4096,
                256, 4096, 512, B_DENSE, 64, 64, EPI_BIAS_RES, lat2_b, nullptr,
                upout + (long)b * 256 * 4096, 4096);

  // ---- deform layer 3 (256 -> 128, 64x64 -> 128x128) ----
  deform_layer(stream, x3, 256, 128, 64, 64, dl3, sc1, sh1, sc2, sh2, om, samp, col, Wp, dcnb, upout);
  for (int b = 0; b < 2; ++b)
    launch_gemm(stream, lat3_w, f2 + (long)b * 256 * 16384, x2 + (long)b * 128 * 16384,
                128, 16384, 256, B_DENSE, 128, 128, EPI_BIAS_RES, lat3_b, nullptr,
                upout + (long)b * 128 * 16384, 16384);

  // ---- aggregation heads + fused multi-scale bilinear sum ----
  for (int b = 0; b < 2; ++b) {
    launch_gemm(stream, a5_w, x5 + (long)b * 1024 * 256,  z5 + (long)b * 128 * 256,
                128, 256, 1024, B_DENSE, 16, 16, EPI_NONE, nullptr, nullptr, nullptr, 256);
    launch_gemm(stream, a4_w, x4 + (long)b * 512 * 1024,  z4 + (long)b * 128 * 1024,
                128, 1024, 512, B_DENSE, 32, 32, EPI_NONE, nullptr, nullptr, nullptr, 1024);
    launch_gemm(stream, a3_w, x3 + (long)b * 256 * 4096,  z3 + (long)b * 128 * 4096,
                128, 4096, 256, B_DENSE, 64, 64, EPI_NONE, nullptr, nullptr, nullptr, 4096);
    launch_gemm(stream, a2_w, x2 + (long)b * 128 * 16384, ybuf + (long)b * 128 * 16384,
                128, 16384, 128, B_DENSE, 128, 128, EPI_NONE, nullptr, nullptr, nullptr, 16384);
    int tot = 128 * 16384;
    fuse_y_kernel<<<(tot + 255) / 256, 256, 0, stream>>>(
        ybuf + (long)b * 128 * 16384,
        z3 + (long)b * 128 * 4096, z4 + (long)b * 128 * 1024, z5 + (long)b * 128 * 256,
        biasP, 128, 128, 128, 64, 64, 32, 32, 16, 16);
  }

  // ---- final 3x3 conv via implicit-im2col WMMA GEMM ----
  for (int b = 0; b < 2; ++b)
    launch_gemm(stream, out_w, ybuf + (long)b * 128 * 16384, out + (long)b * 128 * 16384,
                128, 16384, 128 * 9, B_IM2COL3, 128, 128, EPI_BIAS, out_b, nullptr, nullptr, 16384);
}
